// BasicBlock_13752485281977
// MI455X (gfx1250) — compile-verified, matrix operand order checked by reading, not run on hardware
//
#include <hip/hip_runtime.h>
#include <stdint.h>

typedef _Float16 h16;
typedef __attribute__((ext_vector_type(16))) _Float16 v16h;
typedef __attribute__((ext_vector_type(8)))  float    v8f;
typedef __attribute__((ext_vector_type(4)))  uint32_t v4u;
typedef __attribute__((ext_vector_type(4)))  int      v4i;

#define CI   256
#define CO   256
#define HH   56
#define WW   56
#define NBAT 32
#define PIXI (HH * WW)          // 3136
#define PIX  (NBAT * PIXI)      // 100352 = 784*128 = 392*256
#define KKT  (CI * 9)           // 2304 = 72 * 32
#define KC   32
#define KSTEPS (KKT / KC)       // 72, 8 K-steps per 3x3 tap
#define BN_EPS 1e-5f

#if __has_builtin(__builtin_amdgcn_global_load_async_to_lds_b128)
#define ASYNC_LDS 1
#else
#define ASYNC_LDS 0
#endif

// ---------------- weight min/max -> ternary thresholds -----------------
// mean-centering cancels: w_c < mn_c+third  <=>  w < min(w)+third
__global__ void wstats_kernel(const float* __restrict__ w, float* __restrict__ thr) {
    __shared__ float smn[256], smx[256];
    int t = threadIdx.x;
    float mn = 3.4e38f, mx = -3.4e38f;
    for (int i = t; i < CO * KKT; i += 256) {
        float v = w[i];
        mn = fminf(mn, v);
        mx = fmaxf(mx, v);
    }
    smn[t] = mn; smx[t] = mx;
    __syncthreads();
    for (int s = 128; s > 0; s >>= 1) {
        if (t < s) {
            smn[t] = fminf(smn[t], smn[t + s]);
            smx[t] = fmaxf(smx[t], smx[t + s]);
        }
        __syncthreads();
    }
    if (t == 0) {
        float third = (smx[0] - smn[0]) / 3.0f;
        thr[0] = smn[0] + third;
        thr[1] = smx[0] - third;
    }
}

// ------- ternarize to f16, TAP-MAJOR layout q[o][tap*256 + c] ----------
__global__ void wquant_kernel(const float* __restrict__ w, const float* __restrict__ thr,
                              h16* __restrict__ q) {
    int i = blockIdx.x * 256 + threadIdx.x;        // output (tap-major) index
    if (i < CO * KKT) {
        int o   = i / KKT;
        int r   = i - o * KKT;
        int tap = r >> 8;
        int c   = r & 255;
        float v  = w[(size_t)o * KKT + c * 9 + tap];
        float lo = thr[0], hi = thr[1];
        q[i] = (h16)((v < lo) ? -1.0f : ((v > hi) ? 1.0f : 0.0f));
    }
}

// ------ NCHW f32 -> channel-group-interleaved f16: xh[cg][p][32] -------
__global__ __launch_bounds__(256) void x2h_kernel(const float* __restrict__ x,
                                                  h16* __restrict__ xh) {
    __shared__ h16 tile[32][260];
    const int t  = threadIdx.x;
    const int p0 = blockIdx.x * 256;
    const int cg = blockIdx.y;
    const int p  = p0 + t;
    const int n  = p / PIXI;
    const int hw = p - n * PIXI;
    const float* xs = x + (size_t)(n * CI + cg * 32) * PIXI + hw;
#pragma unroll
    for (int cc = 0; cc < 32; ++cc)
        tile[cc][t] = (h16)xs[(size_t)cc * PIXI];
    __syncthreads();
    h16* d = xh + ((size_t)cg * PIX + p0) * 32;
#pragma unroll
    for (int k = 0; k < 16; ++k) {
        int l = 2 * t + 512 * k;
        union { uint32_t u; h16 hv[2]; } pk;
        pk.hv[0] = tile[l & 31][l >> 5];
        pk.hv[1] = tile[(l + 1) & 31][(l + 1) >> 5];
        *(uint32_t*)&d[l] = pk.u;
    }
}

// ---------------- implicit-GEMM 3x3 conv via WMMA ----------------------
// dst[c_out][pixel]; A = ternary weights (tap-major), B = interleaved acts.
// 256 thr = 8 waves (2M x 4N), tile 128M x 128N, double-buffered LDS.
// Staging: global_load_async_to_lds_b128 (ASYNCcnt) when available,
// else b128 register staging. Fragments: ds_load_b128, conflict-free.
__global__ __launch_bounds__(256) void conv_wmma_kernel(
    const h16* __restrict__ q,     // [CO][KKT] tap-major
    const h16* __restrict__ xin,   // [cg][PIX][32] f16 interleaved
    float* __restrict__ dst)       // [CO][PIX]
{
    __shared__ __align__(16) uint32_t As[2][128][20]; // row=M, 16 uints K + pad->80B
    __shared__ __align__(16) uint32_t Bs[2][128][20]; // row=N, same packing

    const int t    = threadIdx.x;
    const int lane = t & 31;
    const int wid  = t >> 5;
    const int wm   = wid >> 2;            // 0..1
    const int wn   = wid & 3;             // 0..3
    const int Mb   = blockIdx.y * 128;
    const int Nb   = blockIdx.x * 128;
    const int lrow = lane & 15;
    const int lhi  = (lane >> 4) & 1;
    const int uoff = lhi ? 4 : 0;         // +8 halves of K for lanes 16-31

    // A staging: thread -> (row am, half ah); B staging: (pixel nn, half hb)
    const int am = t >> 1;
    const int ah = t & 1;
    const int nn = t & 127;
    const int hb = t >> 7;
    const int Pg = Nb + nn;
    const int n  = Pg / PIXI;
    const int hw = Pg - n * PIXI;
    const int h  = hw / WW;
    const int w  = hw - h * WW;
    const int pixbase = n * PIXI;

    v8f acc[4][2];
#pragma unroll
    for (int a = 0; a < 4; ++a)
#pragma unroll
        for (int b = 0; b < 2; ++b) acc[a][b] = v8f{};

#if ASYNC_LDS
    typedef __attribute__((address_space(1))) v4i gv4i;  // global int4*
    typedef __attribute__((address_space(3))) v4i lv4i;  // LDS int4*
    auto stage_async = [&](int ks, int b) {
        const int k0 = ks * KC;
        gv4i* ga = (gv4i*)(q + (size_t)(Mb + am) * KKT + k0 + ah * 16);
        lv4i* la = (lv4i*)&As[b][am][ah * 8];
        __builtin_amdgcn_global_load_async_to_lds_b128(ga, la, 0, 0);
        __builtin_amdgcn_global_load_async_to_lds_b128(ga + 1, la + 1, 0, 0);
        const int tap = ks >> 3;
        const int t3  = tap / 3;
        const int ih  = h + t3 - 1;
        const int iw  = w + (tap - t3 * 3) - 1;
        const bool ok = ((unsigned)ih < HH) && ((unsigned)iw < WW);
        if (ok) {
            gv4i* gb = (gv4i*)(xin
                + ((size_t)(ks & 7) * PIX + (pixbase + ih * WW + iw)) * 32 + hb * 16);
            lv4i* lb = (lv4i*)&Bs[b][nn][hb * 8];
            __builtin_amdgcn_global_load_async_to_lds_b128(gb, lb, 0, 0);
            __builtin_amdgcn_global_load_async_to_lds_b128(gb + 1, lb + 1, 0, 0);
        } else {
            v4u z = {0u, 0u, 0u, 0u};
            *(v4u*)&Bs[b][nn][hb * 8]     = z;
            *(v4u*)&Bs[b][nn][hb * 8 + 4] = z;
        }
    };
#else
    v4u Av0, Av1, Bv0, Bv1;
    auto stage = [&](int ks) {
        const int k0 = ks * KC;
        const v4u* ga = (const v4u*)(q + (size_t)(Mb + am) * KKT + k0 + ah * 16);
        Av0 = ga[0];
        Av1 = ga[1];
        const int tap = ks >> 3;
        const int t3  = tap / 3;
        const int ih  = h + t3 - 1;
        const int iw  = w + (tap - t3 * 3) - 1;
        const bool ok = ((unsigned)ih < HH) && ((unsigned)iw < WW);
        const v4u* gb = (const v4u*)(xin
            + ((size_t)(ks & 7) * PIX + (pixbase + ih * WW + iw)) * 32 + hb * 16);
        v4u z = {0u, 0u, 0u, 0u};
        Bv0 = ok ? gb[0] : z;
        Bv1 = ok ? gb[1] : z;
    };
    auto commit = [&](int b) {
        *(v4u*)&As[b][am][ah * 8]     = Av0;
        *(v4u*)&As[b][am][ah * 8 + 4] = Av1;
        *(v4u*)&Bs[b][nn][hb * 8]     = Bv0;
        *(v4u*)&Bs[b][nn][hb * 8 + 4] = Bv1;
    };
#endif

#if ASYNC_LDS
    stage_async(0, 0);
    asm volatile("s_wait_asynccnt 0" ::: "memory");
    __syncthreads();
#else
    stage(0);
    commit(0);
    __syncthreads();
#endif

    for (int ks = 0; ks < KSTEPS; ++ks) {
        const int cur = ks & 1;
#if ASYNC_LDS
        if (ks + 1 < KSTEPS) stage_async(ks + 1, 1 - cur);  // async into other buffer
#else
        if (ks + 1 < KSTEPS) stage(ks + 1);
#endif
        if (ks + 4 < KSTEPS)                                 // gfx1250 global_prefetch_b8
            __builtin_prefetch(q + (size_t)(Mb + (t & 127)) * KKT + (ks + 4) * KC, 0, 1);

        // fragments per ISA 16-bit A/B layout: 2 x ds_load_b128 each
        union Frag { v16h h; v4u v[2]; };
        Frag af[4], bf[2];
#pragma unroll
        for (int mt = 0; mt < 4; ++mt) {
            int row = wm * 64 + mt * 16 + lrow;
            af[mt].v[0] = *(const v4u*)&As[cur][row][uoff];      // K 0..7   (+8 hi lanes)
            af[mt].v[1] = *(const v4u*)&As[cur][row][uoff + 8];  // K 16..23 (+8 hi lanes)
        }
#pragma unroll
        for (int nt = 0; nt < 2; ++nt) {
            int row = wn * 32 + nt * 16 + lrow;
            bf[nt].v[0] = *(const v4u*)&Bs[cur][row][uoff];
            bf[nt].v[1] = *(const v4u*)&Bs[cur][row][uoff + 8];
        }
#pragma unroll
        for (int mt = 0; mt < 4; ++mt)
#pragma unroll
            for (int nt = 0; nt < 2; ++nt)
                acc[mt][nt] = __builtin_amdgcn_wmma_f32_16x16x32_f16(
                    false, af[mt].h, false, bf[nt].h,
                    (short)0, acc[mt][nt], false, false);

#if ASYNC_LDS
        asm volatile("s_wait_asynccnt 0" ::: "memory");
#else
        if (ks + 1 < KSTEPS) commit(1 - cur);
#endif
        __syncthreads();
    }

    // writeback: D layout lane = N, vgpr r -> M = r (+8 for hi lanes)
#pragma unroll
    for (int mt = 0; mt < 4; ++mt) {
#pragma unroll
        for (int nt = 0; nt < 2; ++nt) {
            int col = Nb + wn * 32 + nt * 16 + lrow;
#pragma unroll
            for (int r = 0; r < 8; ++r) {
                int row = Mb + wm * 64 + mt * 16 + r + (lhi ? 8 : 0);
                dst[(size_t)row * PIX + col] = acc[mt][nt][r];
            }
        }
    }
}

// ---------------- BN batch stats: one block per channel ----------------
__global__ void bnstats_kernel(const float* __restrict__ src, float* __restrict__ stats) {
    __shared__ float ssum[256], ssq[256];
    int c = blockIdx.x, t = threadIdx.x;
    const float* p = src + (size_t)c * PIX;
    float s = 0.0f, s2 = 0.0f;
    for (int i = t; i < PIX; i += 256) {
        float v = p[i];
        s += v;
        s2 += v * v;
    }
    ssum[t] = s; ssq[t] = s2;
    __syncthreads();
    for (int k = 128; k > 0; k >>= 1) {
        if (t < k) { ssum[t] += ssum[t + k]; ssq[t] += ssq[t + k]; }
        __syncthreads();
    }
    if (t == 0) {
        float m = ssum[0] / (float)PIX;
        float v = ssq[0] / (float)PIX - m * m;
        stats[c * 2]     = m;
        stats[c * 2 + 1] = rsqrtf(v + BN_EPS);
    }
}

// -------- BN + ReLU: [c][p] f32 -> interleaved f16 (for conv2 + res) ----
__global__ __launch_bounds__(256) void bnrelu16_kernel(
    const float* __restrict__ src, const float* __restrict__ stats,
    const float* __restrict__ gamma, const float* __restrict__ beta,
    h16* __restrict__ dst) {
    __shared__ h16 tile[32][260];
    const int t  = threadIdx.x;
    const int p0 = blockIdx.x * 256;
    const int cg = blockIdx.y;
#pragma unroll
    for (int cc = 0; cc < 32; ++cc) {
        int c = cg * 32 + cc;
        float v = src[(size_t)c * PIX + p0 + t];
        v = (v - stats[c * 2]) * stats[c * 2 + 1] * gamma[c] + beta[c];
        tile[cc][t] = (h16)fmaxf(v, 0.0f);
    }
    __syncthreads();
    h16* d = dst + ((size_t)cg * PIX + p0) * 32;
#pragma unroll
    for (int k = 0; k < 16; ++k) {
        int l = 2 * t + 512 * k;
        union { uint32_t u; h16 hv[2]; } pk;
        pk.hv[0] = tile[l & 31][l >> 5];
        pk.hv[1] = tile[(l + 1) & 31][(l + 1) >> 5];
        *(uint32_t*)&d[l] = pk.u;
    }
}

// -------- BN2 + residual + ReLU -> NCHW d_out (LDS un-interleave) -------
__global__ __launch_bounds__(256) void final_kernel(
    const float* __restrict__ conv2raw, const float* __restrict__ stats,
    const float* __restrict__ gamma, const float* __restrict__ beta,
    const h16* __restrict__ out1h, float* __restrict__ out) {
    __shared__ h16 tile[32][260];
    const int t  = threadIdx.x;
    const int p0 = blockIdx.x * 256;
    const int cg = blockIdx.y;
    const h16* s = out1h + ((size_t)cg * PIX + p0) * 32;
#pragma unroll
    for (int k = 0; k < 16; ++k) {
        int l = 2 * t + 512 * k;
        union { uint32_t u; h16 hv[2]; } pk;
        pk.u = *(const uint32_t*)&s[l];
        tile[l & 31][l >> 5]             = pk.hv[0];
        tile[(l + 1) & 31][(l + 1) >> 5] = pk.hv[1];
    }
    __syncthreads();
    const int p  = p0 + t;
    const int n  = p / PIXI;
    const int hw = p - n * PIXI;
#pragma unroll
    for (int cc = 0; cc < 32; ++cc) {
        int c = cg * 32 + cc;
        float v = conv2raw[(size_t)c * PIX + p];
        v = (v - stats[c * 2]) * stats[c * 2 + 1] * gamma[c] + beta[c];
        float r = fmaxf((float)tile[cc][t] + v, 0.0f);
        out[((size_t)n * CO + c) * PIXI + hw] = r;
    }
}

extern "C" void kernel_launch(void* const* d_in, const int* in_sizes, int n_in,
                              void* d_out, int out_size, void* d_ws, size_t ws_size,
                              hipStream_t stream) {
    const float* x  = (const float*)d_in[0];
    const float* w1 = (const float*)d_in[1];
    const float* g1 = (const float*)d_in[2];
    const float* b1 = (const float*)d_in[3];
    const float* w2 = (const float*)d_in[4];
    const float* g2 = (const float*)d_in[5];
    const float* b2 = (const float*)d_in[6];
    float* out = (float*)d_out;

    // workspace (~157 MB): actbuf = x16 for conv1, then out1h (stream-ordered reuse)
    h16*   q1       = (h16*)d_ws;
    h16*   q2       = q1 + (size_t)CO * KKT;
    h16*   actbuf   = q2 + (size_t)CO * KKT;          // CI*PIX halves, interleaved
    float* conv2raw = (float*)(actbuf + (size_t)CI * PIX);
    float* thr      = conv2raw + (size_t)CO * PIX;
    float* stats1   = thr + 8;
    float* stats2   = stats1 + 512;
    float* conv1raw = out;                            // reuse d_out as scratch

    dim3 tgrid(PIX / 256, CI / 32);   // 392 x 8 transpose-style grids
    dim3 cgrid(PIX / 128, CO / 128);  // 784 x 2 conv grid

    wstats_kernel<<<1, 256, 0, stream>>>(w1, thr);
    wstats_kernel<<<1, 256, 0, stream>>>(w2, thr + 2);
    wquant_kernel<<<(CO * KKT + 255) / 256, 256, 0, stream>>>(w1, thr, q1);
    wquant_kernel<<<(CO * KKT + 255) / 256, 256, 0, stream>>>(w2, thr + 2, q2);
    x2h_kernel<<<tgrid, 256, 0, stream>>>(x, actbuf);

    conv_wmma_kernel<<<cgrid, 256, 0, stream>>>(q1, actbuf, conv1raw);
    bnstats_kernel<<<CO, 256, 0, stream>>>(conv1raw, stats1);
    bnrelu16_kernel<<<tgrid, 256, 0, stream>>>(conv1raw, stats1, g1, b1, actbuf);

    conv_wmma_kernel<<<cgrid, 256, 0, stream>>>(q2, actbuf, conv2raw);
    bnstats_kernel<<<CO, 256, 0, stream>>>(conv2raw, stats2);
    final_kernel<<<tgrid, 256, 0, stream>>>(conv2raw, stats2, g2, b2, actbuf, out);
}
